// MolDiffusionNet_3401614098864
// MI455X (gfx1250) — matrix operand order repaired and checked by reading, not compiled, not run on hardware
//
#include <hip/hip_runtime.h>

typedef unsigned short u16;
typedef unsigned int   u32;
typedef __attribute__((ext_vector_type(16))) __bf16 v16bf;
typedef __attribute__((ext_vector_type(8)))  float  v8f;
typedef __attribute__((ext_vector_type(4)))  float  f32x4;

#define NATOMS 4096
#define FDIM   32
#define HID    512

__device__ __forceinline__ u16 f2bf(float f) {
    u32 u = __float_as_uint(f);
    u32 r = u + 0x7FFFu + ((u >> 16) & 1u);   // round-to-nearest-even
    return (u16)(r >> 16);
}
__device__ __forceinline__ float bf2f(u16 h) {
    return __uint_as_float(((u32)h) << 16);
}

// gfx1250 async copy: global -> LDS, tracked by ASYNCcnt
__device__ __forceinline__ void async_copy_b128(u32 lds_off, const void* gaddr) {
    asm volatile("global_load_async_to_lds_b128 %0, %1, off"
                 :: "v"(lds_off), "v"(gaddr) : "memory");
}
__device__ __forceinline__ void wait_async0() {
    asm volatile("s_wait_asynccnt 0x0" ::: "memory");
}
__device__ __forceinline__ u32 lds_off_of(const void* p) {
    return (u32)(size_t)p;   // low 32 bits of generic LDS pointer = LDS byte offset
}

// ---------------------------------------------------------------- deg kernel
__global__ void deg_kernel(const float* __restrict__ adj, float* __restrict__ deginv) {
    int row  = blockIdx.x * 8 + (threadIdx.x >> 5);
    int lane = threadIdx.x & 31;
    const float4* a4 = (const float4*)(adj + (size_t)row * NATOMS);
    float s = 0.f;
    for (int j = lane; j < NATOMS / 4; j += 32) {
        float4 v = a4[j];
        s += v.x + v.y + v.z + v.w;
    }
    #pragma unroll
    for (int off = 16; off > 0; off >>= 1) s += __shfl_xor(s, off, 32);
    if (lane == 0) {
        float d = s < 1.f ? 1.f : s;
        deginv[row] = 1.f / d;
    }
}

// ----------------------------------------------------- f32 -> bf16 (4-wide)
__global__ void cvt_bf16_kernel(const float* __restrict__ in, u16* __restrict__ out, int nquads) {
    int i = blockIdx.x * blockDim.x + threadIdx.x;
    const float4* in4 = (const float4*)in;
    uint2* out2 = (uint2*)out;
    for (; i < nquads; i += gridDim.x * blockDim.x) {
        float4 v = in4[i];
        uint2 o;
        o.x = (u32)f2bf(v.x) | ((u32)f2bf(v.y) << 16);
        o.y = (u32)f2bf(v.z) | ((u32)f2bf(v.w) << 16);
        out2[i] = o;
    }
}

// --------------------------------------- f32 [R,C] -> bf16 transposed [C,R]
__global__ void cvtT_kernel(const float* __restrict__ in, u16* __restrict__ out, int R, int C) {
    __shared__ u16 t[32][33];
    int tr = blockIdx.y * 32, tc = blockIdx.x * 32;
    #pragma unroll
    for (int i = threadIdx.y; i < 32; i += 8)
        t[i][threadIdx.x] = f2bf(in[(size_t)(tr + i) * C + tc + threadIdx.x]);
    __syncthreads();
    #pragma unroll
    for (int i = threadIdx.y; i < 32; i += 8)
        out[(size_t)(tc + i) * R + tr + threadIdx.x] = t[threadIdx.x][i];
}

// -------------------------------------- bf16 [R,C] -> bf16 transposed [C,R]
__global__ void trT_kernel(const u16* __restrict__ in, u16* __restrict__ out, int R, int C) {
    __shared__ u16 t[32][33];
    int tr = blockIdx.y * 32, tc = blockIdx.x * 32;
    #pragma unroll
    for (int i = threadIdx.y; i < 32; i += 8)
        t[i][threadIdx.x] = in[(size_t)(tr + i) * C + tc + threadIdx.x];
    __syncthreads();
    #pragma unroll
    for (int i = threadIdx.y; i < 32; i += 8)
        out[(size_t)(tc + i) * R + tr + threadIdx.x] = t[threadIdx.x][i];
}

// ----------------------------------------------------------- bias1+bias2 add
__global__ void addvec_kernel(const float* a, const float* b, float* o, int n) {
    int i = blockIdx.x * blockDim.x + threadIdx.x;
    if (i < n) o[i] = a[i] + b[i];
}

// ------------------------------------------------------------- WMMA GEMM ---
// C[M,N] = A[M,K](bf16, row major) * Bt[N,K](bf16, row major)^T
// Block tile BM=128 x BN (64 or 32), 8 waves; wave tile 32 x (BN/2).
// EPI 0: store f32 raw            -> outF
// EPI 1: *= rowscale[m], -> bf16  -> outB
// EPI 2: += prevF + bias[n], relu -> bf16 -> outB
// EPI 3: += bias[n], -> f32       -> outF
template<int EPI, int BN>
__global__ __launch_bounds__(256)
void gemm_wmma_kernel(const u16* __restrict__ A, int lda,
                      const u16* __restrict__ Bt, int ldb,
                      int M, int N, int K,
                      float* __restrict__ outF, u16* __restrict__ outB,
                      const float* __restrict__ bias,
                      const float* __restrict__ rowscale,
                      const float* __restrict__ prevF)
{
    constexpr int BM = 128, BK = 32, LDSP = 40;      // padded LDS stride
    constexpr int WNS = BN / 32;                     // 16-col subtiles per wave (2 or 1)
    __shared__ u16 sA[2][BM * LDSP];
    __shared__ u16 sB[2][BN * LDSP];

    const int tid  = threadIdx.x;
    const int wave = tid >> 5, lane = tid & 31;
    const int wm = wave & 3, wn = wave >> 2;         // 4x2 wave grid
    const int g  = lane >> 4, ml = lane & 15;
    const int rowBase = blockIdx.y * BM;
    const int colBase = blockIdx.x * BN;

    v8f acc[2][WNS];
    #pragma unroll
    for (int i = 0; i < 2; ++i)
        #pragma unroll
        for (int j = 0; j < WNS; ++j)
            acc[i][j] = (v8f){0.f,0.f,0.f,0.f,0.f,0.f,0.f,0.f};

    // ISA 7.12.2: 16-bit A 16x32 layout. element e=(vgpr v, half p):
    // lanes 0-15: V0..3 -> K=0..7, V4..7 -> K=16..23 ; lanes 16-31 offset +8
    int ka[16];
    #pragma unroll
    for (int e = 0; e < 16; ++e) {
        int v = e >> 1, p = e & 1;
        ka[e] = (v < 4) ? (8 * g + 2 * v + p) : (16 + 8 * g + 2 * (v - 4) + p);
    }

    // ---- per-thread async-copy assignments (16B segments)
    const int rA  = tid >> 2, segA = tid & 3;        // A rows 0..63 (+64 for 2nd)
    const u16* aSrc0 = A + (size_t)(rowBase + rA)      * lda + segA * 8;
    const u16* aSrc1 = A + (size_t)(rowBase + 64 + rA) * lda + segA * 8;
    const bool bAct = (BN == 64) ? true : (tid < 128);
    const int  nB = tid >> 2, segB = tid & 3;        // B rows 0..BN-1
    const u16* bSrc = Bt + (size_t)(colBase + nB) * ldb + segB * 8;

    u32 aDst0[2], aDst1[2], bDst[2];
    #pragma unroll
    for (int s = 0; s < 2; ++s) {
        aDst0[s] = lds_off_of(&sA[s][rA * LDSP + segA * 8]);
        aDst1[s] = lds_off_of(&sA[s][(64 + rA) * LDSP + segA * 8]);
        bDst[s]  = lds_off_of(&sB[s][nB * LDSP + segB * 8]);
    }

    // prologue: fill buffer 0
    async_copy_b128(aDst0[0], aSrc0);
    async_copy_b128(aDst1[0], aSrc1);
    if (bAct) async_copy_b128(bDst[0], bSrc);
    aSrc0 += BK; aSrc1 += BK; bSrc += BK;

    union Frag { v16bf v; u16 u[16]; };
    auto compute = [&](int s) {
        Frag fa[2], fb[WNS];
        #pragma unroll
        for (int e = 0; e < 16; ++e) {
            #pragma unroll
            for (int i = 0; i < 2; ++i)
                fa[i].u[e] = sA[s][(wm * 32 + i * 16 + ml) * LDSP + ka[e]];
            #pragma unroll
            for (int j = 0; j < WNS; ++j)
                fb[j].u[e] = sB[s][(wn * (BN / 2) + j * 16 + ml) * LDSP + ka[e]];
        }
        #pragma unroll
        for (int i = 0; i < 2; ++i)
            #pragma unroll
            for (int j = 0; j < WNS; ++j)
                acc[i][j] = __builtin_amdgcn_wmma_f32_16x16x32_bf16(
                    false, fa[i].v, false, fb[j].v, (short)0, acc[i][j], false, false);
    };

    const int nIter = K / BK;
    #pragma unroll 2
    for (int it = 0; it < nIter - 1; ++it) {         // steady state: always issue next
        const int s = it & 1;
        wait_async0();          // my async writes into buffer s are complete
        __syncthreads();        // everyone's are; prior readers of buf s^1 done
        async_copy_b128(aDst0[s ^ 1], aSrc0);
        async_copy_b128(aDst1[s ^ 1], aSrc1);
        if (bAct) async_copy_b128(bDst[s ^ 1], bSrc);
        aSrc0 += BK; aSrc1 += BK; bSrc += BK;
        if ((tid & 3) == 0 && it + 2 < nIter)
            __builtin_prefetch(aSrc0, 0, 1);         // global_prefetch_b8, 2 tiles ahead
        compute(s);
    }
    wait_async0();                                   // last tile
    __syncthreads();
    compute((nIter - 1) & 1);

    // ---- epilogue. C/D layout: lane n = ml, VGPR r -> m = 8*g + r
    #pragma unroll
    for (int i = 0; i < 2; ++i) {
        #pragma unroll
        for (int j = 0; j < WNS; ++j) {
            const int n = colBase + wn * (BN / 2) + j * 16 + ml;
            #pragma unroll
            for (int r = 0; r < 8; ++r) {
                int m = rowBase + wm * 32 + i * 16 + g * 8 + r;
                float val = acc[i][j][r];
                size_t off = (size_t)m * N + n;
                if (EPI == 0) {
                    outF[off] = val;
                } else if (EPI == 1) {
                    outB[off] = f2bf(val * rowscale[m]);
                } else if (EPI == 2) {
                    float t = val + prevF[off] + bias[n];
                    outB[off] = f2bf(t > 0.f ? t : 0.f);
                } else {
                    outF[off] = val + bias[n];
                }
            }
        }
    }
}

// ----------------------------------------------- bond head vectors a,b (Nx4)
__global__ void bond_ab_kernel(const u16* __restrict__ h2b, const float* __restrict__ wbond,
                               float* __restrict__ avec, float* __restrict__ bvec)
{
    int row  = blockIdx.x * 8 + (threadIdx.x >> 5);
    int lane = threadIdx.x & 31;
    float aa[4] = {0,0,0,0}, bb[4] = {0,0,0,0};
    for (int k = lane; k < HID; k += 32) {
        float h = bf2f(h2b[(size_t)row * HID + k]);
        #pragma unroll
        for (int c = 0; c < 4; ++c) {
            aa[c] += h * wbond[c * (2 * HID) + k];
            bb[c] += h * wbond[c * (2 * HID) + HID + k];
        }
    }
    #pragma unroll
    for (int c = 0; c < 4; ++c) {
        #pragma unroll
        for (int off = 16; off > 0; off >>= 1) {
            aa[c] += __shfl_xor(aa[c], off, 32);
            bb[c] += __shfl_xor(bb[c], off, 32);
        }
    }
    if (lane < 4) {
        avec[row * 4 + lane] = aa[lane];
        bvec[row * 4 + lane] = bb[lane];
    }
}

// ------------------------------- bond_logits: nontemporal float4 stream-out
__global__ void bond_pair_kernel(const float* __restrict__ avec, const float* __restrict__ bvec,
                                 const float* __restrict__ bbond, float* __restrict__ out)
{
    size_t idx = (size_t)blockIdx.x * blockDim.x + threadIdx.x;  // over N*N pairs
    int i = (int)(idx >> 12);
    int j = (int)(idx & (NATOMS - 1));
    f32x4 r = {0.f, 0.f, 0.f, 0.f};
    if (i != j) {
        int lo = i < j ? i : j;   // upper-triangle source index
        int hi = i < j ? j : i;
        f32x4 av = ((const f32x4*)avec)[lo];
        f32x4 bv = ((const f32x4*)bvec)[hi];
        f32x4 bb = *((const f32x4*)bbond);
        r = av + bv + bb;
    }
    __builtin_nontemporal_store(r, (f32x4*)out + idx);   // 268 MB > L2, keep NT
}

// ===========================================================================
extern "C" void kernel_launch(void* const* d_in, const int* in_sizes, int n_in,
                              void* d_out, int out_size, void* d_ws, size_t ws_size,
                              hipStream_t stream)
{
    const float* x        = (const float*)d_in[0];
    const float* adj      = (const float*)d_in[1];
    const float* w_self1  = (const float*)d_in[2];
    const float* b_self1  = (const float*)d_in[3];
    const float* w_neigh1 = (const float*)d_in[4];
    const float* b_neigh1 = (const float*)d_in[5];
    const float* w_self2  = (const float*)d_in[6];
    const float* b_self2  = (const float*)d_in[7];
    const float* w_neigh2 = (const float*)d_in[8];
    const float* b_neigh2 = (const float*)d_in[9];
    const float* w_atom   = (const float*)d_in[10];
    const float* b_atom   = (const float*)d_in[11];
    const float* w_bond   = (const float*)d_in[12];
    const float* b_bond   = (const float*)d_in[13];

    // ---- workspace carve-out
    char* w = (char*)d_ws;
    size_t off = 0;
    auto alloc = [&](size_t bytes) -> char* {
        char* p = w + off;
        off += (bytes + 255) & ~(size_t)255;
        return p;
    };
    u16*   adjb   = (u16*)  alloc((size_t)NATOMS * NATOMS * 2);
    u16*   xb     = (u16*)  alloc((size_t)NATOMS * FDIM * 2);
    u16*   xbT    = (u16*)  alloc((size_t)FDIM * NATOMS * 2);
    u16*   aggb   = (u16*)  alloc((size_t)NATOMS * FDIM * 2);
    u16*   h1b    = (u16*)  alloc((size_t)NATOMS * HID * 2);
    u16*   h1bT   = (u16*)  alloc((size_t)HID * NATOMS * 2);
    u16*   agg2b  = (u16*)  alloc((size_t)NATOMS * HID * 2);
    u16*   h2b    = (u16*)  alloc((size_t)NATOMS * HID * 2);
    float* tmpF   = (float*)alloc((size_t)NATOMS * HID * 4);
    u16*   ws1b   = (u16*)  alloc((size_t)HID * FDIM * 2);
    u16*   wn1b   = (u16*)  alloc((size_t)HID * FDIM * 2);
    u16*   ws2b   = (u16*)  alloc((size_t)HID * HID * 2);
    u16*   wn2b   = (u16*)  alloc((size_t)HID * HID * 2);
    u16*   watomb = (u16*)  alloc((size_t)FDIM * HID * 2);
    float* deginv = (float*)alloc((size_t)NATOMS * 4);
    float* bias1  = (float*)alloc((size_t)HID * 4);
    float* bias2  = (float*)alloc((size_t)HID * 4);
    float* avec   = (float*)alloc((size_t)NATOMS * 4 * 4);
    float* bvec   = (float*)alloc((size_t)NATOMS * 4 * 4);
    (void)ws_size; (void)n_in; (void)in_sizes; (void)out_size;

    float* atom_out = (float*)d_out;                      // 4096*32
    float* bond_out = (float*)d_out + (size_t)NATOMS * FDIM;

    // ---- 1/deg
    deg_kernel<<<NATOMS / 8, 256, 0, stream>>>(adj, deginv);

    // ---- bf16 conversions (+ transposes for the B operands)
    auto cvt = [&](const float* src, u16* dst, size_t n) {
        int nq = (int)(n / 4);
        int blocks = (nq + 255) / 256; if (blocks > 16384) blocks = 16384;
        cvt_bf16_kernel<<<blocks, 256, 0, stream>>>(src, dst, nq);
    };
    cvt(adj, adjb, (size_t)NATOMS * NATOMS);
    cvt(x, xb, (size_t)NATOMS * FDIM);
    cvt(w_self1, ws1b, (size_t)HID * FDIM);
    cvt(w_neigh1, wn1b, (size_t)HID * FDIM);
    cvt(w_self2, ws2b, (size_t)HID * HID);
    cvt(w_neigh2, wn2b, (size_t)HID * HID);
    cvt(w_atom, watomb, (size_t)FDIM * HID);
    cvtT_kernel<<<dim3(FDIM / 32, NATOMS / 32), dim3(32, 8), 0, stream>>>(x, xbT, NATOMS, FDIM);

    addvec_kernel<<<1, HID, 0, stream>>>(b_self1, b_neigh1, bias1, HID);
    addvec_kernel<<<1, HID, 0, stream>>>(b_self2, b_neigh2, bias2, HID);

    dim3 blk(256);
    // ---- layer 1 ----------------------------------------------------------
    // agg = (adj @ x) / deg                     [M=4096, N=32, K=4096]
    gemm_wmma_kernel<1, 32><<<dim3(FDIM / 32, NATOMS / 128), blk, 0, stream>>>(
        adjb, NATOMS, xbT, NATOMS, NATOMS, FDIM, NATOMS,
        nullptr, aggb, nullptr, deginv, nullptr);
    // tmpF = x @ w_self1^T                      [M=4096, N=512, K=32]
    gemm_wmma_kernel<0, 64><<<dim3(HID / 64, NATOMS / 128), blk, 0, stream>>>(
        xb, FDIM, ws1b, FDIM, NATOMS, HID, FDIM,
        tmpF, nullptr, nullptr, nullptr, nullptr);
    // h1 = relu(tmpF + agg @ w_neigh1^T + bias1) -> bf16
    gemm_wmma_kernel<2, 64><<<dim3(HID / 64, NATOMS / 128), blk, 0, stream>>>(
        aggb, FDIM, wn1b, FDIM, NATOMS, HID, FDIM,
        nullptr, h1b, bias1, nullptr, tmpF);
    // h1^T for the big aggregation GEMM
    trT_kernel<<<dim3(HID / 32, NATOMS / 32), dim3(32, 8), 0, stream>>>(h1b, h1bT, NATOMS, HID);

    // ---- layer 2 ----------------------------------------------------------
    // agg2 = (adj @ h1) / deg                   [M=4096, N=512, K=4096]  (big one)
    gemm_wmma_kernel<1, 64><<<dim3(HID / 64, NATOMS / 128), blk, 0, stream>>>(
        adjb, NATOMS, h1bT, NATOMS, NATOMS, HID, NATOMS,
        nullptr, agg2b, nullptr, deginv, nullptr);
    // tmpF = h1 @ w_self2^T                     [M=4096, N=512, K=512]
    gemm_wmma_kernel<0, 64><<<dim3(HID / 64, NATOMS / 128), blk, 0, stream>>>(
        h1b, HID, ws2b, HID, NATOMS, HID, HID,
        tmpF, nullptr, nullptr, nullptr, nullptr);
    // h2 = relu(tmpF + agg2 @ w_neigh2^T + bias2) -> bf16
    gemm_wmma_kernel<2, 64><<<dim3(HID / 64, NATOMS / 128), blk, 0, stream>>>(
        agg2b, HID, wn2b, HID, NATOMS, HID, HID,
        nullptr, h2b, bias2, nullptr, tmpF);

    // ---- heads ------------------------------------------------------------
    // atom_logits = h2 @ w_atom^T + b_atom -> f32 d_out
    gemm_wmma_kernel<3, 32><<<dim3(FDIM / 32, NATOMS / 128), blk, 0, stream>>>(
        h2b, HID, watomb, HID, NATOMS, FDIM, HID,
        atom_out, nullptr, b_atom, nullptr, nullptr);

    // bond head vectors, then the big streaming broadcast-add
    bond_ab_kernel<<<NATOMS / 8, 256, 0, stream>>>(h2b, w_bond, avec, bvec);
    bond_pair_kernel<<<(int)(((size_t)NATOMS * NATOMS) / 256), 256, 0, stream>>>(
        avec, bvec, b_bond, bond_out);
}
